// local_correlation_softmax_6253472383688
// MI455X (gfx1250) — matrix operand order repaired and checked by reading, not compiled, not run on hardware
//
#include <hip/hip_runtime.h>
#include <hip/hip_bf16.h>

// Problem constants (fixed by the reference setup).
#define BATCH 2
#define CH    128
#define HH    96
#define WW    128
#define RAD   4
#define WIN   9          // 2*RAD+1
#define KWIN  81         // WIN*WIN
#define TILE  16         // pixels per wave along x
#define WAVES 4          // waves per block
#define HWSZ  (HH * WW)

typedef __attribute__((ext_vector_type(16))) __bf16 v16bf;
typedef __attribute__((ext_vector_type(8)))  __bf16 v8bf;
typedef __attribute__((ext_vector_type(8)))  float  v8f;

static __device__ __forceinline__ __bf16 f32_to_bf16_rne(float x) {
    union { float f; unsigned u; } c; c.f = x;
    unsigned r = c.u + 0x7FFFu + ((c.u >> 16) & 1u);   // round-to-nearest-even
    unsigned short h = (unsigned short)(r >> 16);
    return __builtin_bit_cast(__bf16, h);
}

static __device__ __forceinline__ v16bf cat16(v8bf lo, v8bf hi) {
    return __builtin_shufflevector(lo, hi, 0, 1, 2, 3, 4, 5, 6, 7,
                                           8, 9, 10, 11, 12, 13, 14, 15);
}

// Wave-private LDS ordering: LDS ops of one wave complete in order (DScnt);
// the fence only pins compiler ordering + inserts the minimal wait.
static __device__ __forceinline__ void wave_lds_fence() {
    __builtin_amdgcn_fence(__ATOMIC_ACQ_REL, "wavefront");
}

// ---------------------------------------------------------------------------
// Prepass: f32 [b,c,h,w] -> bf16 channel-last [b,h,w,c] for both features.
// ---------------------------------------------------------------------------
__global__ __launch_bounds__(256)
void bf16_chlast_prepass(const float* __restrict__ f0,
                         const float* __restrict__ f1,
                         unsigned short* __restrict__ wsOut) {
    const int pix   = blockIdx.x * blockDim.x + threadIdx.x;  // [0, 2*B*HW)
    const int total = BATCH * HWSZ;
    const bool second = pix >= total;
    const int p  = second ? pix - total : pix;
    const int b  = p / HWSZ;
    const int p2 = p % HWSZ;
    const float* src = (second ? f1 : f0) + (size_t)b * CH * HWSZ + p2;
    __bf16* dst = (__bf16*)wsOut + (size_t)pix * CH;

    for (int c = 0; c < CH; c += 8) {
        v8bf t;
#pragma unroll
        for (int i = 0; i < 8; ++i)
            t[i] = f32_to_bf16_rne(src[(size_t)(c + i) * HWSZ]);
        *(v8bf*)(dst + c) = t;
    }
}

// ---------------------------------------------------------------------------
// Main kernel: per-wave 16-pixel tile; software-pipelined window rows.
// ---------------------------------------------------------------------------
template <bool FAST>
__global__ __launch_bounds__(WAVES * 32)
void local_corr_softmax_wmma_kernel(const float* __restrict__ f0,
                                    const float* __restrict__ f1,
                                    const unsigned short* __restrict__ t0,
                                    const unsigned short* __restrict__ t1,
                                    float* __restrict__ out) {
    __shared__ float Gs[WAVES][TILE][32];
    __shared__ float Cs[WAVES][TILE][KWIN];

    const int lane = threadIdx.x & 31;
    const int wid  = threadIdx.x >> 5;
    const int tile = blockIdx.x * WAVES + wid;      // grid sized exactly

    const int tilesPerRow = WW / TILE;              // 8
    const int b   = tile / (HH * tilesPerRow);
    const int r2  = tile % (HH * tilesPerRow);
    const int y   = r2 / tilesPerRow;
    const int x0  = (r2 % tilesPerRow) * TILE;

    const int m    = lane & 15;   // pixel row of A / column index of B
    const int half = lane >> 4;

    // Clamped sample columns for the two B tiles (q = m and q = 16+m).
    const int xa = min(max(x0 - RAD + m, 0), WW - 1);
    const int xb = min(max(x0 - RAD + 16 + m, 0), WW - 1);

    // ---- A fragments: feature0 tile (M=16 pixels, K=128 channels) ----
    v16bf A[4];
    if constexpr (FAST) {
        const __bf16* F0T = (const __bf16*)t0 +
            ((size_t)b * HWSZ + (size_t)y * WW + (size_t)(x0 + m)) * CH;
#pragma unroll
        for (int c4 = 0; c4 < 4; ++c4) {
            const int kc = c4 * 32;
            A[c4] = cat16(*(const v8bf*)(F0T + kc + half * 16),
                          *(const v8bf*)(F0T + kc + half * 16 + 8));
        }
    } else {
        const float* F0 = f0 + (size_t)b * CH * HWSZ;
#pragma unroll
        for (int c4 = 0; c4 < 4; ++c4) {
            const int kc = c4 * 32;
#pragma unroll
            for (int v = 0; v < 8; ++v) {
                const int ks = kc + (half * 16) + (v * 2);
                const float* p = F0 + (size_t)ks * HWSZ + (size_t)y * WW + (x0 + m);
                A[c4][2 * v]     = f32_to_bf16_rne(p[0]);
                A[c4][2 * v + 1] = f32_to_bf16_rne(p[HWSZ]);
            }
        }
    }

    // Issue one window row's worth of B-fragment loads (consumed later).
    auto loadB = [&](int j, v16bf (&B0)[4], v16bf (&B1)[4]) {
        const int yyc = min(max(y + j - RAD, 0), HH - 1);
        if constexpr (FAST) {
            const __bf16* rowT = (const __bf16*)t1 +
                ((size_t)b * HWSZ + (size_t)yyc * WW) * CH;
            const __bf16* colA = rowT + (size_t)xa * CH;
            const __bf16* colB = rowT + (size_t)xb * CH;
            if (j + 1 < WIN) {   // prefetch row after next (global_prefetch_b8)
                const int yn = min(max(y + j + 1 - RAD, 0), HH - 1);
                const __bf16* nrow = (const __bf16*)t1 +
                    ((size_t)b * HWSZ + (size_t)yn * WW + (size_t)xa) * CH;
                __builtin_prefetch(nrow, 0, 1);
            }
#pragma unroll
            for (int c4 = 0; c4 < 4; ++c4) {
                const int kb = c4 * 32 + half * 16;
                B0[c4] = cat16(*(const v8bf*)(colA + kb),
                               *(const v8bf*)(colA + kb + 8));
                B1[c4] = cat16(*(const v8bf*)(colB + kb),
                               *(const v8bf*)(colB + kb + 8));
            }
        } else {
            const float* F1 = f1 + (size_t)b * CH * HWSZ;
#pragma unroll
            for (int c4 = 0; c4 < 4; ++c4) {
                const int kb = c4 * 32 + half * 16;
#pragma unroll
                for (int v = 0; v < 8; ++v) {
                    const float* pc = F1 + (size_t)(kb + 2 * v) * HWSZ +
                                      (size_t)yyc * WW;
                    B0[c4][2 * v]     = f32_to_bf16_rne(pc[xa]);
                    B0[c4][2 * v + 1] = f32_to_bf16_rne(pc[HWSZ + xa]);
                    B1[c4][2 * v]     = f32_to_bf16_rne(pc[xb]);
                    B1[c4][2 * v + 1] = f32_to_bf16_rne(pc[HWSZ + xb]);
                }
            }
        }
    };

    const float scale = 0.08838834764831845f;  // 1/sqrt(128)

    v16bf B0buf[2][4], B1buf[2][4];
    loadB(0, B0buf[0], B1buf[0]);

#pragma unroll
    for (int j = 0; j < WIN; ++j) {            // window row dy = j - RAD
        const int cur = j & 1;
        const int nxt = cur ^ 1;
        if (j + 1 < WIN)                        // pipeline: next row's loads
            loadB(j + 1, B0buf[nxt], B1buf[nxt]);

        v8f acc0 = {};
        v8f acc1 = {};
#pragma unroll
        for (int c4 = 0; c4 < 4; ++c4) {
            acc0 = __builtin_amdgcn_wmma_f32_16x16x32_bf16(
                       false, A[c4], false, B0buf[cur][c4], (short)0, acc0,
                       false, false);
            acc1 = __builtin_amdgcn_wmma_f32_16x16x32_bf16(
                       false, A[c4], false, B1buf[cur][c4], (short)0, acc1,
                       false, false);
        }

        // Stage G (16x32) in wave-private LDS for the cross-lane band gather.
        wave_lds_fence();
#pragma unroll
        for (int v = 0; v < 8; ++v) {
            Gs[wid][v + 8 * half][m]      = acc0[v];
            Gs[wid][v + 8 * half][16 + m] = acc1[v];
        }
        wave_lds_fence();

        // corr[p, jx] = G[p, p+jx] (OOB sample -> logit 0, like the reference).
        const int yy = y + j - RAD;
        if (lane < TILE) {
            const int p = lane;
#pragma unroll
            for (int jx = 0; jx < WIN; ++jx) {
                const int sx = x0 + p + jx - RAD;
                const bool valid = (yy >= 0) && (yy < HH) && (sx >= 0) && (sx < WW);
                const float g = Gs[wid][p][p + jx];
                Cs[wid][p][j * WIN + jx] = valid ? g * scale : 0.0f;
            }
        }
    }

    // ---- softmax over 81 + weighted coords -> flow (Cs rows lane-private) ----
    if (lane < TILE) {
        const int p = lane;
        float mx = -1e30f;
        for (int k = 0; k < KWIN; ++k) mx = fmaxf(mx, Cs[wid][p][k]);
        float s = 0.0f;
        for (int k = 0; k < KWIN; ++k) {
            const float e = __expf(Cs[wid][p][k] - mx);
            Cs[wid][p][k] = e;
            s += e;
        }
        const float inv = 1.0f / s;
        float cx = 0.0f, cy = 0.0f;
        int k = 0;
        for (int a = 0; a < WIN; ++a) {
            for (int q = 0; q < WIN; ++q, ++k) {
                const float pr = Cs[wid][p][k] * inv;
                Cs[wid][p][k] = pr;
                cx += pr * (float)(x0 + p + q - RAD);   // unclamped sample coords
                cy += pr * (float)(y + a - RAD);
            }
        }
        const size_t hw   = (size_t)HWSZ;
        const size_t base = (size_t)b * 2 * hw + (size_t)y * WW + (size_t)(x0 + p);
        out[base]      = cx - (float)(x0 + p);   // flow x
        out[base + hw] = cy - (float)y;          // flow y
    }
    wave_lds_fence();

    // ---- match_prob [b, h*w, 81] appended after flow ----
    // Lane (p, half) writes k in [half*40, half*40+41); k=40 double-written
    // with the identical value (benign, deterministic).
    const size_t flowElems = (size_t)BATCH * 2 * HWSZ;
    const size_t probBase  = flowElems +
        ((size_t)b * HWSZ + (size_t)y * WW + (size_t)(x0 + m)) * KWIN;
    for (int kk = 0; kk < 41; ++kk) {
        const int k = kk + half * 40;
        out[probBase + k] = Cs[wid][m][k];
    }
}

extern "C" void kernel_launch(void* const* d_in, const int* in_sizes, int n_in,
                              void* d_out, int out_size, void* d_ws, size_t ws_size,
                              hipStream_t stream) {
    (void)in_sizes; (void)n_in; (void)out_size;
    const float* f0 = (const float*)d_in[0];
    const float* f1 = (const float*)d_in[1];
    // d_in[2] = local_radius (always 4 here; compile-time constant).
    float* out = (float*)d_out;

    const int totalTiles = BATCH * HH * (WW / TILE);     // 1536 waves
    dim3 grid(totalTiles / WAVES);                        // 384 blocks
    dim3 block(WAVES * 32);                               // 128 threads = 4 waves

    const size_t needWs = (size_t)2 * BATCH * HWSZ * CH * sizeof(unsigned short);
    if (d_ws != nullptr && ws_size >= needWs) {
        unsigned short* wsT = (unsigned short*)d_ws;
        const int prepThreads = 2 * BATCH * HWSZ;         // 49152
        bf16_chlast_prepass<<<dim3(prepThreads / 256), dim3(256), 0, stream>>>(
            f0, f1, wsT);
        const unsigned short* t0 = wsT;
        const unsigned short* t1 = wsT + (size_t)BATCH * HWSZ * CH;
        local_corr_softmax_wmma_kernel<true><<<grid, block, 0, stream>>>(
            f0, f1, t0, t1, out);
    } else {
        local_corr_softmax_wmma_kernel<false><<<grid, block, 0, stream>>>(
            f0, f1, nullptr, nullptr, out);
    }
}
